// CRSDCell_25804163514995
// MI455X (gfx1250) — compile-verified, matrix operand order checked by reading, not run on hardware
//
#include <hip/hip_runtime.h>
#include <math.h>

typedef __attribute__((ext_vector_type(16))) __bf16 v16bf;
typedef __attribute__((ext_vector_type(8)))  float  v8f;
typedef __attribute__((ext_vector_type(4)))  unsigned uv4;
typedef __attribute__((__vector_size__(16))) int v4i;

#define B_SZ 8
#define T_SZ 8192
#define DX 256
#define DH 512
#define DK 128
#define DV 128
#define RR 512
#define NSLOTS 256
#define TOPK 8
#define LN_EPS 1e-5f

#define HAS_ASYNC_LDS __has_builtin(__builtin_amdgcn_global_load_async_to_lds_b128)

static __device__ __forceinline__ __bf16 f2bf(float f) {
    unsigned u = __builtin_bit_cast(unsigned, f);
    u += 0x7fffu + ((u >> 16) & 1u);                 // round-to-nearest-even
    unsigned short h = (unsigned short)(u >> 16);
    return __builtin_bit_cast(__bf16, h);
}
static __device__ __forceinline__ float sigm(float x) { return 1.0f / (1.0f + expf(-x)); }

// Stage a contiguous bf16 tile from global into LDS. Async-to-LDS when available.
static __device__ __forceinline__ void stage_tile(const __bf16* __restrict__ gsrc,
                                                  __bf16* __restrict__ ldst,
                                                  int bytes, int tid) {
#if HAS_ASYNC_LDS
    const char* g = (const char*)gsrc;
    char* l = (char*)ldst;
    for (int off = tid * 16; off < bytes; off += 256 * 16) {
        __builtin_amdgcn_global_load_async_to_lds_b128(
            (__attribute__((address_space(1))) v4i*)(g + off),
            (__attribute__((address_space(3))) v4i*)(l + off), 0, 0);
    }
#else
    const uv4* g = (const uv4*)gsrc;
    uv4* l = (uv4*)ldst;
    int n = bytes >> 4;
    for (int i = tid; i < n; i += 256) l[i] = g[i];
#endif
}
static __device__ __forceinline__ void stage_wait() {
#if HAS_ASYNC_LDS
#if __has_builtin(__builtin_amdgcn_s_wait_asynccnt)
    __builtin_amdgcn_s_wait_asynccnt(0);
#else
    asm volatile("s_wait_asynccnt 0" ::: "memory");
#endif
#endif
    __syncthreads();
}

// Load a 16-element bf16 fragment per ISA 16-bit A/B layout:
// caller passes pointer at [row][k0 + kbase]; elems 0..7 = K+0..7, 8..15 = K+16..23
static __device__ __forceinline__ v16bf load_frag(const __bf16* p) {
    v16bf r;
#pragma unroll
    for (int i = 0; i < 8; ++i) { r[i] = p[i]; r[i + 8] = p[16 + i]; }
    return r;
}
static __device__ __forceinline__ v8f wmma_bf(v16bf a, v16bf b, v8f c) {
    return __builtin_amdgcn_wmma_f32_16x16x32_bf16(false, a, false, b, (short)0, c, false, false);
}

// ---------------- prep: f32 -> bf16 conversion, zero accumulator ----------------
__global__ void k_cvt(const float* __restrict__ src, __bf16* __restrict__ dst, int n) {
    int i = blockIdx.x * blockDim.x + threadIdx.x;
    if (i < n) dst[i] = f2bf(src[i]);
}
__global__ void k_zero(float* __restrict__ p, int n) {
    int i = blockIdx.x * blockDim.x + threadIdx.x;
    if (i < n) p[i] = 0.0f;
}

// ---------------- stage 1: r_seq = LN((1-alpha)*tanh(x @ Wx^T + b)) ----------------
template <bool CACHED>
__global__ __launch_bounds__(256)
void k_reservoir(const float* __restrict__ x, const __bf16* __restrict__ xBf,
                 const __bf16* __restrict__ WxBf,
                 const float* __restrict__ Wx_b, const float* __restrict__ res_alpha,
                 const float* __restrict__ rn_g, const float* __restrict__ rn_b,
                 float* __restrict__ r_out, __bf16* __restrict__ rBf)
{
    __shared__ __align__(16) __bf16 xA[16 * DX];
    __shared__ float rbuf[16 * RR];
    const int tid = threadIdx.x;
    const int gr0 = blockIdx.x * 16;
    if (CACHED) {
        stage_tile(xBf + (size_t)gr0 * DX, xA, 16 * DX * 2, tid);
        stage_wait();
    } else {
        const float* xsrc = x + (size_t)gr0 * DX;
        for (int i = tid; i < 16 * DX; i += 256) xA[i] = f2bf(xsrc[i]);
        __syncthreads();
    }

    const int lane = tid & 31, wv = tid >> 5;
    const int rA = lane & 15, kof = (lane >> 4) << 3, rbase = (lane >> 4) << 3;
    v8f z = {0,0,0,0,0,0,0,0};
    v8f acc[4] = {z, z, z, z};
    for (int k0 = 0; k0 < DX; k0 += 32) {
        v16bf a = load_frag(&xA[rA * DX + k0 + kof]);
#pragma unroll
        for (int j = 0; j < 4; ++j) {
            int n0 = (wv * 4 + j) * 16;
            v16bf b = load_frag(&WxBf[(size_t)(n0 + rA) * DX + k0 + kof]);
            acc[j] = wmma_bf(a, b, acc[j]);
        }
    }
#pragma unroll
    for (int j = 0; j < 4; ++j) {
        int col = (wv * 4 + j) * 16 + (lane & 15);
        float oma = 1.0f - sigm(res_alpha[col >> 7]);
        float bias = Wx_b[col];
#pragma unroll
        for (int v = 0; v < 8; ++v)
            rbuf[(v + rbase) * RR + col] = oma * tanhf(acc[j][v] + bias);
    }
    __syncthreads();
    // LayerNorm per row (16 threads/row, shfl reduce within 16-lane groups)
    const int row = tid >> 4, sub = tid & 15;
    float s = 0.f, sq = 0.f;
#pragma unroll
    for (int j = 0; j < 32; ++j) { float v = rbuf[row * RR + sub + 16 * j]; s += v; sq += v * v; }
#pragma unroll
    for (int m = 1; m < 16; m <<= 1) { s += __shfl_xor(s, m, 32); sq += __shfl_xor(sq, m, 32); }
    float mean = s * (1.0f / RR);
    float rstd = rsqrtf(sq * (1.0f / RR) - mean * mean + LN_EPS);
    float* dst = r_out + (size_t)gr0 * RR;
#pragma unroll
    for (int j = 0; j < 32; ++j) {
        int col = sub + 16 * j;
        float v = (rbuf[row * RR + col] - mean) * rstd * rn_g[col] + rn_b[col];
        dst[row * RR + col] = v;
        if (CACHED) rBf[(size_t)gr0 * RR + row * RR + col] = f2bf(v);
    }
}

// ---------------- stage 2: qsum += sum_t l2norm([r|x] @ key_w^T + key_b) ----------------
template <bool CACHED>
__global__ __launch_bounds__(256)
void k_query(const float* __restrict__ r_seq, const float* __restrict__ x,
             const __bf16* __restrict__ rBf, const __bf16* __restrict__ xBf,
             const __bf16* __restrict__ keyBf, const float* __restrict__ key_b,
             float* __restrict__ qsum)
{
    __shared__ __align__(16) __bf16 rT[16 * RR];
    __shared__ __align__(16) __bf16 xT[16 * DX];
    __shared__ float kbuf[16 * DK];
    __shared__ float nrm[16];
    const int tid = threadIdx.x;
    const int gr0 = blockIdx.x * 16;
    const int b = gr0 / T_SZ;
    if (CACHED) {
        stage_tile(rBf + (size_t)gr0 * RR, rT, 16 * RR * 2, tid);
        stage_tile(xBf + (size_t)gr0 * DX, xT, 16 * DX * 2, tid);
        stage_wait();
    } else {
        const float* rsrc = r_seq + (size_t)gr0 * RR;
        const float* xsrc = x + (size_t)gr0 * DX;
        for (int i = tid; i < 16 * RR; i += 256) rT[i] = f2bf(rsrc[i]);
        for (int i = tid; i < 16 * DX; i += 256) xT[i] = f2bf(xsrc[i]);
        __syncthreads();
    }

    const int lane = tid & 31, wv = tid >> 5;
    const int rA = lane & 15, kof = (lane >> 4) << 3, rbase = (lane >> 4) << 3;
    const int LD = RR + DX;
    v8f acc = {0,0,0,0,0,0,0,0};
    for (int k0 = 0; k0 < RR; k0 += 32) {
        v16bf a = load_frag(&rT[rA * RR + k0 + kof]);
        v16bf bb = load_frag(&keyBf[(size_t)(wv * 16 + rA) * LD + k0 + kof]);
        acc = wmma_bf(a, bb, acc);
    }
    for (int k0 = 0; k0 < DX; k0 += 32) {
        v16bf a = load_frag(&xT[rA * DX + k0 + kof]);
        v16bf bb = load_frag(&keyBf[(size_t)(wv * 16 + rA) * LD + RR + k0 + kof]);
        acc = wmma_bf(a, bb, acc);
    }
    int col = wv * 16 + (lane & 15);
    float bias = key_b[col];
#pragma unroll
    for (int v = 0; v < 8; ++v) kbuf[(v + rbase) * DK + col] = acc[v] + bias;
    __syncthreads();

    int row = tid >> 4, sub = tid & 15;
    float sq = 0.f;
#pragma unroll
    for (int j = 0; j < 8; ++j) { float v = kbuf[row * DK + sub + 16 * j]; sq += v * v; }
#pragma unroll
    for (int m = 1; m < 16; m <<= 1) sq += __shfl_xor(sq, m, 32);
    if (sub == 0) nrm[row] = fmaxf(sqrtf(sq), 1e-8f);
    __syncthreads();

    int ccol = tid & 127, half = tid >> 7;
    float part = 0.f;
#pragma unroll
    for (int r8 = 0; r8 < 8; ++r8) {
        int rr = half * 8 + r8;
        part += kbuf[rr * DK + ccol] / nrm[rr];
    }
    atomicAdd(&qsum[b * DK + ccol], part);
}

// ---------------- stage 3: episodic top-k + hebbian read -> 0.5*h_mem ----------------
__global__ __launch_bounds__(256)
void k_memory(const float* __restrict__ qsum, const float* __restrict__ mem_keys,
              const float* __restrict__ mem_vals, const float* __restrict__ heb,
              const float* __restrict__ rm_w, const float* __restrict__ rm_b,
              const float* __restrict__ mix_logit, float* __restrict__ hmem_half)
{
    __shared__ float q[DK], qn[DK], sim[NSLOTS], vcomb[DK];
    __shared__ float topv[TOPK], wtk[TOPK];
    __shared__ int   topi[TOPK];
    const int tid = threadIdx.x;
    const int b = blockIdx.x;
    if (tid < DK) q[tid] = qsum[b * DK + tid] * (1.0f / T_SZ);
    __syncthreads();
    if (tid == 0) {
        float s = 0.f;
        for (int d = 0; d < DK; ++d) s += q[d] * q[d];
        float inv = 1.0f / fmaxf(sqrtf(s), 1e-8f);
        for (int d = 0; d < DK; ++d) qn[d] = q[d] * inv;
    }
    __syncthreads();
    {
        const float* mk = mem_keys + ((size_t)b * NSLOTS + tid) * DK;
        float dot = 0.f, nn = 0.f;
        for (int d = 0; d < DK; ++d) { float m = mk[d]; dot += qn[d] * m; nn += m * m; }
        sim[tid] = dot / fmaxf(sqrtf(nn), 1e-8f);
    }
    __syncthreads();
    if (tid == 0) {
        for (int j = 0; j < TOPK; ++j) {
            float best = -1e30f; int bi = 0;
            for (int m = 0; m < NSLOTS; ++m) if (sim[m] > best) { best = sim[m]; bi = m; }
            topv[j] = best; topi[j] = bi; sim[bi] = -1e30f;
        }
        float mx = topv[0], se = 0.f;
        for (int j = 0; j < TOPK; ++j) { wtk[j] = expf(topv[j] - mx); se += wtk[j]; }
        for (int j = 0; j < TOPK; ++j) wtk[j] /= se;
    }
    __syncthreads();
    if (tid < DK) {
        float vh = 0.f;
        for (int j = 0; j < TOPK; ++j)
            vh += wtk[j] * mem_vals[((size_t)b * NSLOTS + topi[j]) * DK + tid];
        float vb = 0.f;
        for (int k = 0; k < DK; ++k) vb += q[k] * heb[((size_t)b * DK + k) * DV + tid];
        float mix = sigm(mix_logit[0]);
        vcomb[tid] = mix * vh + (1.0f - mix) * vb;
    }
    __syncthreads();
    for (int h = tid; h < DH; h += 256) {
        float s = rm_b[h];
        for (int d = 0; d < DV; ++d) s += rm_w[(size_t)h * DV + d] * vcomb[d];
        hmem_half[b * DH + h] = 0.5f * s;
    }
}

// ---------------- stage 4: h_tilde = gelu(r@B^T + x@U^T + b); hs = blend + h_mem ----------------
template <bool CACHED>
__global__ __launch_bounds__(256)
void k_core(const float* __restrict__ r_seq, const float* __restrict__ x,
            const __bf16* __restrict__ rBf, const __bf16* __restrict__ xBf,
            const __bf16* __restrict__ BBf, const __bf16* __restrict__ UBf,
            const __bf16* __restrict__ PBf, const float* __restrict__ U_b,
            const float* __restrict__ fftn_g, const float* __restrict__ fftn_b,
            const float* __restrict__ fftp_b, const float* __restrict__ fft_mix,
            const float* __restrict__ hmem_half, float* __restrict__ hs_out)
{
    __shared__ __align__(16) char smem[16 * RR * 2 + 16 * DX * 2 + 16 * DH * 4]; // 57344 B
    __bf16* rT = (__bf16*)smem;                              // 16 x 512 bf16
    __bf16* xT = (__bf16*)(smem + 16 * RR * 2);              // 16 x 256 bf16
    float*  ht = (float*)(smem + 16 * RR * 2 + 16 * DX * 2); // 16 x 512 f32
    __bf16* hn = (__bf16*)smem;                              // aliases rT after GEMM1

    const int tid = threadIdx.x;
    const int gr0 = blockIdx.x * 16;
    const int b = gr0 / T_SZ;
    if (CACHED) {
        stage_tile(rBf + (size_t)gr0 * RR, rT, 16 * RR * 2, tid);
        stage_tile(xBf + (size_t)gr0 * DX, xT, 16 * DX * 2, tid);
        stage_wait();
    } else {
        const float* rsrc = r_seq + (size_t)gr0 * RR;
        const float* xsrc = x + (size_t)gr0 * DX;
        for (int i = tid; i < 16 * RR; i += 256) rT[i] = f2bf(rsrc[i]);
        for (int i = tid; i < 16 * DX; i += 256) xT[i] = f2bf(xsrc[i]);
        __syncthreads();
    }

    const int lane = tid & 31, wv = tid >> 5;
    const int rA = lane & 15, kof = (lane >> 4) << 3, rbase = (lane >> 4) << 3;
    v8f z = {0,0,0,0,0,0,0,0};
    v8f acc[4] = {z, z, z, z};
    for (int k0 = 0; k0 < RR; k0 += 32) {                     // Bh
        v16bf a = load_frag(&rT[rA * RR + k0 + kof]);
#pragma unroll
        for (int j = 0; j < 4; ++j) {
            int n0 = (wv * 4 + j) * 16;
            v16bf bb = load_frag(&BBf[(size_t)(n0 + rA) * RR + k0 + kof]);
            acc[j] = wmma_bf(a, bb, acc[j]);
        }
    }
    for (int k0 = 0; k0 < DX; k0 += 32) {                     // Uh
        v16bf a = load_frag(&xT[rA * DX + k0 + kof]);
#pragma unroll
        for (int j = 0; j < 4; ++j) {
            int n0 = (wv * 4 + j) * 16;
            v16bf bb = load_frag(&UBf[(size_t)(n0 + rA) * DX + k0 + kof]);
            acc[j] = wmma_bf(a, bb, acc[j]);
        }
    }
#pragma unroll
    for (int j = 0; j < 4; ++j) {
        int col = (wv * 4 + j) * 16 + (lane & 15);
        float bias = U_b[col];
#pragma unroll
        for (int v = 0; v < 8; ++v) {
            float val = acc[j][v] + bias;
            ht[(v + rbase) * DH + col] = 0.5f * val * (1.0f + erff(val * 0.70710678118f)); // exact gelu
        }
    }
    __syncthreads();
    {   // LayerNorm -> bf16 hn (aliases rT; all rT reads completed before barrier)
        int row = tid >> 4, sub = tid & 15;
        float s = 0.f, sq = 0.f;
#pragma unroll
        for (int j = 0; j < 32; ++j) { float v = ht[row * DH + sub + 16 * j]; s += v; sq += v * v; }
#pragma unroll
        for (int m = 1; m < 16; m <<= 1) { s += __shfl_xor(s, m, 32); sq += __shfl_xor(sq, m, 32); }
        float mean = s * (1.0f / DH);
        float rstd = rsqrtf(sq * (1.0f / DH) - mean * mean + LN_EPS);
#pragma unroll
        for (int j = 0; j < 32; ++j) {
            int col = sub + 16 * j;
            hn[row * DH + col] = f2bf((ht[row * DH + col] - mean) * rstd * fftn_g[col] + fftn_b[col]);
        }
    }
    __syncthreads();
    v8f acc2[4] = {z, z, z, z};
    for (int k0 = 0; k0 < DH; k0 += 32) {                     // fftp GEMM
        v16bf a = load_frag(&hn[rA * DH + k0 + kof]);
#pragma unroll
        for (int j = 0; j < 4; ++j) {
            int n0 = (wv * 4 + j) * 16;
            v16bf bb = load_frag(&PBf[(size_t)(n0 + rA) * DH + k0 + kof]);
            acc2[j] = wmma_bf(a, bb, acc2[j]);
        }
    }
    float sgl = sigm(fft_mix[0]);
    float* dst = hs_out + (size_t)gr0 * DH;
#pragma unroll
    for (int j = 0; j < 4; ++j) {
        int col = (wv * 4 + j) * 16 + (lane & 15);
        float pb = fftp_b[col];
        float hm = hmem_half[b * DH + col];
#pragma unroll
        for (int v = 0; v < 8; ++v) {
            int row = v + rbase;
            float fft_out = sgl * acc2[j][v] + pb;            // (s*h_norm)@P^T + b
            dst[row * DH + col] = (1.0f - sgl) * ht[row * DH + col] + sgl * fft_out + hm;
        }
    }
}

extern "C" void kernel_launch(void* const* d_in, const int* in_sizes, int n_in,
                              void* d_out, int out_size, void* d_ws, size_t ws_size,
                              hipStream_t stream)
{
    (void)in_sizes; (void)n_in; (void)out_size;
    const float* x        = (const float*)d_in[0];
    const float* Wx_w     = (const float*)d_in[1];
    const float* Wx_b     = (const float*)d_in[2];
    /* d_in[3] Wh_w unused: h_prev == 0 */
    const float* res_alpha= (const float*)d_in[4];
    /* d_in[5] A_scale unused: h_prev == 0 */
    const float* B_w      = (const float*)d_in[6];
    const float* U_w      = (const float*)d_in[7];
    const float* U_b      = (const float*)d_in[8];
    const float* fftn_g   = (const float*)d_in[9];
    const float* fftn_b   = (const float*)d_in[10];
    const float* fftp_w   = (const float*)d_in[11];
    const float* fftp_b   = (const float*)d_in[12];
    const float* fft_mix  = (const float*)d_in[13];
    const float* mix_logit= (const float*)d_in[14];
    const float* key_w    = (const float*)d_in[15];
    const float* key_b    = (const float*)d_in[16];
    /* d_in[17] val_w, d_in[18] val_b unused: v never consumed */
    const float* rm_w     = (const float*)d_in[19];
    const float* rm_b     = (const float*)d_in[20];
    const float* rn_g     = (const float*)d_in[21];
    const float* rn_b     = (const float*)d_in[22];
    const float* mem_keys = (const float*)d_in[23];
    const float* mem_vals = (const float*)d_in[24];
    const float* heb      = (const float*)d_in[25];

    char* ws = (char*)d_ws;
    size_t off = 0;
    __bf16* WxBf = (__bf16*)(ws + off); off += (size_t)RR * DX * 2;
    __bf16* BBf  = (__bf16*)(ws + off); off += (size_t)DH * RR * 2;
    __bf16* UBf  = (__bf16*)(ws + off); off += (size_t)DH * DX * 2;
    __bf16* PBf  = (__bf16*)(ws + off); off += (size_t)DH * DH * 2;
    __bf16* KBf  = (__bf16*)(ws + off); off += (size_t)DK * (RR + DX) * 2;
    float*  qsum = (float*)(ws + off);  off += (size_t)B_SZ * DK * 4;
    float*  hmem = (float*)(ws + off);  off += (size_t)B_SZ * DH * 4;
    __bf16* xBf  = (__bf16*)(ws + off); off += (size_t)B_SZ * T_SZ * DX * 2;  // 33.5 MB
    __bf16* rBf  = (__bf16*)(ws + off); off += (size_t)B_SZ * T_SZ * RR * 2;  // 67 MB
    const bool cached = (ws_size >= off);

    k_cvt<<<(RR * DX + 255) / 256, 256, 0, stream>>>(Wx_w, WxBf, RR * DX);
    k_cvt<<<(DH * RR + 255) / 256, 256, 0, stream>>>(B_w, BBf, DH * RR);
    k_cvt<<<(DH * DX + 255) / 256, 256, 0, stream>>>(U_w, UBf, DH * DX);
    k_cvt<<<(DH * DH + 255) / 256, 256, 0, stream>>>(fftp_w, PBf, DH * DH);
    k_cvt<<<(DK * (RR + DX) + 255) / 256, 256, 0, stream>>>(key_w, KBf, DK * (RR + DX));
    k_zero<<<(B_SZ * DK + 255) / 256, 256, 0, stream>>>(qsum, B_SZ * DK);

    float* hs_out = (float*)d_out;                             // (B,T,DH)
    float* r_out  = hs_out + (size_t)B_SZ * T_SZ * DH;         // (B,T,R)
    const int tiles = (B_SZ * T_SZ) / 16;                      // 4096

    if (cached) {
        k_cvt<<<(B_SZ * T_SZ * DX + 255) / 256, 256, 0, stream>>>(x, xBf, B_SZ * T_SZ * DX);
        k_reservoir<true><<<tiles, 256, 0, stream>>>(x, xBf, WxBf, Wx_b, res_alpha, rn_g, rn_b,
                                                     r_out, rBf);
        k_query<true><<<tiles, 256, 0, stream>>>(r_out, x, rBf, xBf, KBf, key_b, qsum);
        k_memory<<<B_SZ, 256, 0, stream>>>(qsum, mem_keys, mem_vals, heb, rm_w, rm_b, mix_logit, hmem);
        k_core<true><<<tiles, 256, 0, stream>>>(r_out, x, rBf, xBf, BBf, UBf, PBf, U_b,
                                                fftn_g, fftn_b, fftp_b, fft_mix, hmem, hs_out);
    } else {
        k_reservoir<false><<<tiles, 256, 0, stream>>>(x, nullptr, WxBf, Wx_b, res_alpha, rn_g, rn_b,
                                                      r_out, nullptr);
        k_query<false><<<tiles, 256, 0, stream>>>(r_out, x, nullptr, nullptr, KBf, key_b, qsum);
        k_memory<<<B_SZ, 256, 0, stream>>>(qsum, mem_keys, mem_vals, heb, rm_w, rm_b, mix_logit, hmem);
        k_core<false><<<tiles, 256, 0, stream>>>(r_out, x, nullptr, nullptr, BBf, UBf, PBf, U_b,
                                                 fftn_g, fftn_b, fftp_b, fft_mix, hmem, hs_out);
    }
}
